// SparseAttentionEdgePredictorLayer_74586402063284
// MI455X (gfx1250) — compile-verified, matrix operand order checked by reading, not run on hardware
//
#include <hip/hip_runtime.h>
#include <hip/hip_bf16.h>

typedef __attribute__((ext_vector_type(16))) _Float16 v16h;
typedef __attribute__((ext_vector_type(8)))  float    v8f;

#define NN   4096
#define DD   128

// ---------------------------------------------------------------------------
// Unified fragment loader (CDNA5 ISA 7.12.2, wave32, 16-bit A 16x32 layout).
// LDS tile is stored "row per fragment lane": row index = matrix M (for A) or
// matrix N (for B stored transposed); columns = K.
//   lane holds row m = lane&15; half-wave selects K-octet:
//   VGPR v, slot s: k = 16*(v>=4) + 8*(lane>=16) + 2*(v&3) + s
// Per lane this is two contiguous 8-half (16B) chunks -> 2x ds_load_b128.
// C/D (16x16 f32): VGPR r: lanes0-15 -> (m=r, n=lane), lanes16-31 -> (m=r+8).
// ---------------------------------------------------------------------------
__device__ __forceinline__ v16h load_frag(const _Float16* lds, int ld,
                                          int row0, int k0, int lane) {
  int m   = row0 + (lane & 15);
  int khi = (lane >> 4) << 3;           // 0 or 8
  v16h a;
#pragma unroll
  for (int v = 0; v < 8; ++v) {
    int kb = k0 + ((v >= 4) ? 16 : 0) + khi + ((v & 3) << 1);
    a[2 * v]     = lds[m * ld + kb];
    a[2 * v + 1] = lds[m * ld + kb + 1];
  }
  return a;
}

// ---------------------------------------------------------------------------
// Kernel 1: S[n][m] = exp( sum_d (wq[d]*f[d][n]) * (wk[d]*f[d][m]) )
// One block -> 128x128 output tile.  K = 128 (whole feature dim in LDS).
// 8 waves; each wave owns a 64x32 region = 4x2 WMMA tiles.
// ---------------------------------------------------------------------------
__global__ __launch_bounds__(256)
void s_kernel(const float* __restrict__ f, const float* __restrict__ wq,
              const float* __restrict__ wk, float* __restrict__ S,
              _Float16* __restrict__ Sh) {
  __shared__ _Float16 As[128][DD + 8];   // Q^T tile: As[n][d]
  __shared__ _Float16 Bs[128][DD + 8];   // K^T tile: Bs[m][d]  (B stored as B^T)

  const int tid = threadIdx.x;
  const int n0  = blockIdx.y * 128;
  const int m0  = blockIdx.x * 128;

  // Stage + scale + cvt to f16.  2 d-rows per pass, 128 lanes per row.
  const int col  = tid & 127;
  const int drow = tid >> 7;            // 0..1
  for (int d = drow; d < DD; d += 2) {
    float qv = wq[d] * f[d * NN + n0 + col];
    float kv = wk[d] * f[d * NN + m0 + col];
    As[col][d] = (_Float16)qv;
    Bs[col][d] = (_Float16)kv;
  }
  __syncthreads();

  const int wave = tid >> 5;
  const int lane = tid & 31;
  const int wr   = wave >> 2;           // 0..1 -> row offset wr*64
  const int wc   = wave & 3;            // 0..3 -> col offset wc*32

  v8f acc[4][2] = {};
#pragma unroll
  for (int kk = 0; kk < DD; kk += 32) {
    v16h bf[2];
#pragma unroll
    for (int j = 0; j < 2; ++j)
      bf[j] = load_frag(&Bs[0][0], DD + 8, wc * 32 + j * 16, kk, lane);
#pragma unroll
    for (int i = 0; i < 4; ++i) {
      v16h af = load_frag(&As[0][0], DD + 8, wr * 64 + i * 16, kk, lane);
#pragma unroll
      for (int j = 0; j < 2; ++j)
        acc[i][j] = __builtin_amdgcn_wmma_f32_16x16x32_f16(
            false, af, false, bf[j], (short)0, acc[i][j], false, false);
    }
  }

  // Epilogue: exp, write f32 (final S numerator) and f16 (GEMM2 operand).
#pragma unroll
  for (int i = 0; i < 4; ++i) {
#pragma unroll
    for (int j = 0; j < 2; ++j) {
      int mbase = n0 + wr * 64 + i * 16;
      int nbase = m0 + wc * 32 + j * 16;
      int cn    = nbase + (lane & 15);
      int moff  = (lane >= 16) ? 8 : 0;
#pragma unroll
      for (int r = 0; r < 8; ++r) {
        int   cm = mbase + r + moff;
        float e  = __expf(acc[i][j][r]);
        S[cm * NN + cn]  = e;
        Sh[cm * NN + cn] = (_Float16)e;
      }
    }
  }
}

// ---------------------------------------------------------------------------
// Kernel 2: denom = neighbors @ S   (4096^3 f16 GEMM, f32 accumulate)
// 128x128 tile per block, BK=32, LDS double-buffered, one barrier/iter.
// Both tiles in LDS are [row][k] with row = fragment-lane index.
// ---------------------------------------------------------------------------
#define BK   32
#define LDK  (BK + 8)                    // 40 halves = 80B rows (16B aligned)

__global__ __launch_bounds__(256)
void denom_kernel(const float* __restrict__ Nb, const _Float16* __restrict__ Sh,
                  float* __restrict__ denom) {
  __shared__ _Float16 As2[2][128][LDK];  // neighbors tile [m][k]
  __shared__ _Float16 Bs2[2][128][LDK];  // S^T tile       [n][k]

  const int tid = threadIdx.x;
  const int i0  = blockIdx.y * 128;
  const int j0  = blockIdx.x * 128;

  const int wave = tid >> 5;
  const int lane = tid & 31;
  const int wr   = wave >> 2;
  const int wc   = wave & 3;

  const int acol = tid & 31;            // k within tile for A
  const int arow = tid >> 5;            // 0..7, rows stride 8
  const int bcol = tid & 127;           // n within tile for B
  const int brow = tid >> 7;            // 0..1, rows stride 2

  auto stage = [&](int kt, int b) {
    const int k0 = kt * BK;
#pragma unroll
    for (int t = 0; t < 16; ++t) {
      int r = arow + 8 * t;
      As2[b][r][acol] = (_Float16)Nb[(i0 + r) * NN + k0 + acol];
    }
#pragma unroll
    for (int t = 0; t < 16; ++t) {
      int r = brow + 2 * t;
      Bs2[b][bcol][r] = Sh[(k0 + r) * NN + j0 + bcol];   // transposed store
    }
  };

  v8f acc[4][2] = {};

  stage(0, 0);
  __syncthreads();

  const int NT = NN / BK;
  for (int kt = 0; kt < NT; ++kt) {
    const int cur = kt & 1;
    if (kt + 1 < NT) stage(kt + 1, cur ^ 1);

    v16h bf[2];
#pragma unroll
    for (int j = 0; j < 2; ++j)
      bf[j] = load_frag(&Bs2[cur][0][0], LDK, wc * 32 + j * 16, 0, lane);
#pragma unroll
    for (int i = 0; i < 4; ++i) {
      v16h af = load_frag(&As2[cur][0][0], LDK, wr * 64 + i * 16, 0, lane);
#pragma unroll
      for (int j = 0; j < 2; ++j)
        acc[i][j] = __builtin_amdgcn_wmma_f32_16x16x32_f16(
            false, af, false, bf[j], (short)0, acc[i][j], false, false);
    }
    __syncthreads();
  }

#pragma unroll
  for (int i = 0; i < 4; ++i) {
#pragma unroll
    for (int j = 0; j < 2; ++j) {
      int mbase = i0 + wr * 64 + i * 16;
      int nbase = j0 + wc * 32 + j * 16;
      int cn    = nbase + (lane & 15);
      int moff  = (lane >= 16) ? 8 : 0;
#pragma unroll
      for (int r = 0; r < 8; ++r)
        denom[(mbase + r + moff) * NN + cn] = acc[i][j][r];
    }
  }
}

// ---------------------------------------------------------------------------
// Kernel 3: S[i][j] /= denom[j][i]  (LDS 32x33 tile transpose)
// ---------------------------------------------------------------------------
__global__ __launch_bounds__(256)
void norm_kernel(float* __restrict__ S, const float* __restrict__ denom) {
  __shared__ float t[32][33];
  const int i0 = blockIdx.y * 32;
  const int j0 = blockIdx.x * 32;
  const int tx = threadIdx.x & 31;
  const int ty = threadIdx.x >> 5;      // 0..7

#pragma unroll
  for (int r = ty; r < 32; r += 8)
    t[tx][r] = denom[(j0 + r) * NN + i0 + tx];  // t[i][j] = denom[j0+j][i0+i]
  __syncthreads();
#pragma unroll
  for (int r = ty; r < 32; r += 8) {
    int idx = (i0 + r) * NN + j0 + tx;
    S[idx] = S[idx] / t[r][tx];
  }
}

extern "C" void kernel_launch(void* const* d_in, const int* in_sizes, int n_in,
                              void* d_out, int out_size, void* d_ws, size_t ws_size,
                              hipStream_t stream) {
  const float* f         = (const float*)d_in[0];   // [128, 4096]
  const float* neighbors = (const float*)d_in[1];   // [4096, 4096]
  const float* wq        = (const float*)d_in[2];   // [128]
  const float* wk        = (const float*)d_in[3];   // [128]

  float*    S     = (float*)d_out;                                  // 64 MB
  _Float16* Sh    = (_Float16*)d_ws;                                // 32 MB
  float*    denom = (float*)((char*)d_ws + (size_t)NN * NN * 2);    // 64 MB

  dim3 blk(256);
  dim3 g_gemm(NN / 128, NN / 128);      // 32 x 32
  dim3 g_norm(NN / 32, NN / 32);        // 128 x 128

  s_kernel<<<g_gemm, blk, 0, stream>>>(f, wq, wk, S, Sh);
  denom_kernel<<<g_gemm, blk, 0, stream>>>(neighbors, Sh, denom);
  norm_kernel<<<g_norm, blk, 0, stream>>>(S, denom);
}
